// Relation_Encoder_61168924229762
// MI455X (gfx1250) — matrix operand order repaired and checked by reading, not compiled, not run on hardware
//
#include <hip/hip_runtime.h>
#include <hip/hip_bf16.h>

typedef __bf16 bf16;
typedef __attribute__((ext_vector_type(16))) __bf16 bf16x16;
typedef __attribute__((ext_vector_type(8)))  float   floatx8;

#define B_  32
#define N_  300
#define D_  1024
#define KK_ 64     // NONGT
#define H_  16
#define PED_ 64
#define DG_ 64

// ---------------------------------------------------------------------------
// Fragment loaders (ISA 7.12.2 layouts, wave32)
// A 16x32 bf16: lane l -> m = l&15 ; elem i -> k = ((i>>3)<<4) + (i&7) + 8*(l>>4)
// B 32x16 bf16: lane l -> n = l&15 ; elem i -> k = ((i>>3)<<4) + (i&7) + 8*(l>>4)
// C/D 16x16 f32: vgpr j, lane l -> m = j + 8*(l>>4), n = l&15
// ---------------------------------------------------------------------------
__device__ __forceinline__ bf16x16 frag_a_rowmajor(const bf16* base, int ld) {
  int l = threadIdx.x & 31;
  int m = l & 15;
  int kh = (l >> 4) << 3;
  bf16x16 r;
#pragma unroll
  for (int i = 0; i < 16; i += 2) {
    int k = ((i >> 3) << 4) + (i & 7) + kh;
    r[i]     = base[m * ld + k];
    r[i + 1] = base[m * ld + k + 1];
  }
  return r;
}

__device__ __forceinline__ bf16x16 frag_b_rows(const bf16* base, long ld) {
  int l = threadIdx.x & 31;
  bf16x16 r;
#pragma unroll
  for (int i = 0; i < 16; ++i) r[i] = base[(long)(((i >> 3) << 4) + (i & 7) + 8 * (l >> 4)) * ld + (l & 15)];
  return r;
}

// ---------------------------------------------------------------------------
// CDNA5 async global->LDS copy (ASYNCcnt-tracked DMA, no VGPR staging).
//   lds  : wave-relative LDS byte address (low 32 bits of generic pointer)
//   gptr : 64-bit global address
// ---------------------------------------------------------------------------
__device__ __forceinline__ void async_ld16(const bf16* lds_generic, const bf16* gptr) {
  unsigned lds = (unsigned)(unsigned long long)lds_generic;
  unsigned long long ga = (unsigned long long)gptr;
  asm volatile("global_load_async_to_lds_b128 %0, %1, off"
               :: "v"(lds), "v"(ga) : "memory");
}
__device__ __forceinline__ void wait_async0() {
  asm volatile("s_wait_asynccnt 0" ::: "memory");
}

// ---------------------------------------------------------------------------
// Elementwise f32 -> bf16
// ---------------------------------------------------------------------------
__global__ void __launch_bounds__(256) cvt_bf16_kernel(const float* __restrict__ in,
                                                       bf16* __restrict__ out, long n) {
  long i = (long)blockIdx.x * blockDim.x + threadIdx.x;
  if (i < n) out[i] = (bf16)in[i];
}

// ---------------------------------------------------------------------------
// Generalized batched transpose + convert:
//   z = blockIdx.z -> zb = z/Hdiv, zh = z%Hdiv
//   in  slab: in  + zb*iSB + zh*iSH, (R,C) f32 with row stride ldin
//   out slab: out + zb*oSB + zh*oSH, (C,R) bf16 with row stride ldout
// ---------------------------------------------------------------------------
__global__ void __launch_bounds__(256) transpose_cvt_kernel(
    const float* __restrict__ in, bf16* __restrict__ out,
    int R, int C, long ldin, long ldout,
    long iSB, long iSH, long oSB, long oSH, int Hdiv) {
  __shared__ float tile[32][33];
  int z = blockIdx.z;
  int zb = z / Hdiv, zh = z % Hdiv;
  in  += (long)zb * iSB + (long)zh * iSH;
  out += (long)zb * oSB + (long)zh * oSH;
  int c0 = blockIdx.x * 32, r0 = blockIdx.y * 32;
  int tx = threadIdx.x & 31, ty = threadIdx.x >> 5;  // ty 0..7
#pragma unroll
  for (int rr = ty; rr < 32; rr += 8) {
    int r = r0 + rr, c = c0 + tx;
    if (r < R && c < C) tile[rr][tx] = in[(long)r * ldin + c];
  }
  __syncthreads();
#pragma unroll
  for (int cc = ty; cc < 32; cc += 8) {
    int c = c0 + cc, r = r0 + tx;
    if (c < C && r < R) out[(long)c * ldout + r] = (bf16)tile[tx][cc];
  }
}

// ---------------------------------------------------------------------------
// Big-tile bf16 WMMA GEMM (non-batched): C = (A @ Bt + bias) * scale
//   M,N multiples of 128; K multiple of 64 (iters even, >=2). 256 thr = 8 waves.
//   Block tile 128m x 128n; wave tile 32m x 64n (waves on 4x2 grid).
//   Async global->LDS double buffering (ASYNCcnt), parity unrolled 2x so all
//   LDS bases are compile-time constants; one s_wait_asynccnt + barrier/slice.
// ---------------------------------------------------------------------------
__global__ void __launch_bounds__(256) gemm_bf16_big_kernel(
    const bf16* __restrict__ A, const bf16* __restrict__ Bt,
    const float* __restrict__ bias,
    float* __restrict__ Cf, bf16* __restrict__ Cb,
    int M, int N, int K, float scale) {
  __shared__ __align__(16) bf16 sA[2][128 * 32];
  __shared__ __align__(16) bf16 sB[2][32 * 128];
  int tid = threadIdx.x;
  int w = tid >> 5, l = tid & 31;
  int m0 = blockIdx.y * 128, n0 = blockIdx.x * 128;
  int wm = (w >> 1) * 32, wn = (w & 1) * 64;

  floatx8 acc[2][4];
#pragma unroll
  for (int s = 0; s < 2; ++s)
#pragma unroll
    for (int nt = 0; nt < 4; ++nt) acc[s][nt] = floatx8{0.f};

  auto issueSlice = [&](int k0, int buf) {
#pragma unroll
    for (int ii = 0; ii < 2; ++ii) {
      int f = tid + ii * 256;              // 0..511 int4 (8 bf16 each)
      int row = f >> 2, ko = (f & 3) * 8;
      async_ld16(&sA[buf][f * 8], A + (long)(m0 + row) * K + k0 + ko);
    }
#pragma unroll
    for (int ii = 0; ii < 2; ++ii) {
      int f = tid + ii * 256;
      int kr = f >> 4, no = (f & 15) * 8;
      async_ld16(&sB[buf][f * 8], Bt + (long)(k0 + kr) * N + n0 + no);
    }
  };
  auto computeStep = [&](int buf) {
    const bf16* cA = sA[buf];
    const bf16* cB = sB[buf];
    bf16x16 a0 = frag_a_rowmajor(cA + (wm) * 32, 32);
    bf16x16 a1 = frag_a_rowmajor(cA + (wm + 16) * 32, 32);
#pragma unroll
    for (int nt = 0; nt < 4; ++nt) {
      bf16x16 b = frag_b_rows(cB + wn + nt * 16, 128);
      acc[0][nt] = __builtin_amdgcn_wmma_f32_16x16x32_bf16(
          false, a0, false, b, (short)0, acc[0][nt], false, false);
      acc[1][nt] = __builtin_amdgcn_wmma_f32_16x16x32_bf16(
          false, a1, false, b, (short)0, acc[1][nt], false, false);
    }
  };

  const int iters = K >> 5;                 // even, >= 2
  issueSlice(0, 0);
  wait_async0();
  __syncthreads();
  for (int it = 0; it + 2 < iters; it += 2) {
    issueSlice((it + 1) << 5, 1);           // DMA into buf1
    computeStep(0);                         // WMMAs from buf0 (latency hidden)
    wait_async0();
    __syncthreads();
    issueSlice((it + 2) << 5, 0);           // DMA into buf0
    computeStep(1);                         // WMMAs from buf1
    wait_async0();
    __syncthreads();
  }
  issueSlice((iters - 1) << 5, 1);
  computeStep(0);
  wait_async0();
  __syncthreads();
  computeStep(1);

  int half = l >> 4, lc = l & 15;
#pragma unroll
  for (int s = 0; s < 2; ++s)
#pragma unroll
    for (int nt = 0; nt < 4; ++nt) {
      int n = n0 + wn + nt * 16 + lc;
      float bv = bias ? bias[n] : 0.f;
#pragma unroll
      for (int j = 0; j < 8; ++j) {
        int m = m0 + wm + s * 16 + j + 8 * half;
        float vv = (acc[s][nt][j] + bv) * scale;
        long idx = (long)m * N + n;
        if (Cf) Cf[idx] = vv;
        if (Cb) Cb[idx] = (bf16)vv;
      }
    }
}

// ---------------------------------------------------------------------------
// Small batched bf16 WMMA GEMM (M,N multiples of 64; K mult of 64):
//   C[z] = (A[z] @ Bt[z] + bias) * scale, z -> (zb,zh) with separate strides.
//   128 threads = 4 waves, block tile 64x64, wave tile 16x64.
//   Same async double-buffer scheme as the big kernel.
// ---------------------------------------------------------------------------
__global__ void __launch_bounds__(128) gemm_bf16_kernel(
    const bf16* __restrict__ A,  long aSB, long aSH,
    const bf16* __restrict__ Bt, long bSB, long bSH,
    const float* __restrict__ bias,
    float* __restrict__ Cf, bf16* __restrict__ Cb, long cSB, long cSH,
    int M, int N, int K, int Hdiv, float scale) {
  __shared__ __align__(16) bf16 sA[2][64 * 32];
  __shared__ __align__(16) bf16 sB[2][32 * 64];
  int z = blockIdx.z;
  int zb = z / Hdiv, zh = z % Hdiv;
  A  += (long)zb * aSB + (long)zh * aSH;
  Bt += (long)zb * bSB + (long)zh * bSH;
  long cOff = (long)zb * cSB + (long)zh * cSH;

  int tid = threadIdx.x;
  int w = tid >> 5, l = tid & 31;
  int m0 = blockIdx.y * 64;
  int n0 = blockIdx.x * 64;

  floatx8 acc[4] = {floatx8{0.f}, floatx8{0.f}, floatx8{0.f}, floatx8{0.f}};

  auto issueSlice = [&](int k0, int buf) {
#pragma unroll
    for (int ii = 0; ii < 2; ++ii) {
      int f = tid + ii * 128;
      int row = f >> 2, ko = (f & 3) * 8;
      async_ld16(&sA[buf][f * 8], A + (long)(m0 + row) * K + k0 + ko);
    }
#pragma unroll
    for (int ii = 0; ii < 2; ++ii) {
      int f = tid + ii * 128;
      int kr = f >> 3, no = (f & 7) * 8;
      async_ld16(&sB[buf][f * 8], Bt + (long)(k0 + kr) * N + n0 + no);
    }
  };
  auto computeStep = [&](int buf) {
    const bf16* cA = sA[buf];
    const bf16* cB = sB[buf];
    bf16x16 a = frag_a_rowmajor(cA + (w * 16) * 32, 32);
#pragma unroll
    for (int nt = 0; nt < 4; ++nt) {
      bf16x16 b = frag_b_rows(cB + nt * 16, 64);
      acc[nt] = __builtin_amdgcn_wmma_f32_16x16x32_bf16(
          false, a, false, b, (short)0, acc[nt], false, false);
    }
  };

  const int iters = K >> 5;                 // even, >= 2
  issueSlice(0, 0);
  wait_async0();
  __syncthreads();
  for (int it = 0; it + 2 < iters; it += 2) {
    issueSlice((it + 1) << 5, 1);
    computeStep(0);
    wait_async0();
    __syncthreads();
    issueSlice((it + 2) << 5, 0);
    computeStep(1);
    wait_async0();
    __syncthreads();
  }
  issueSlice((iters - 1) << 5, 1);
  computeStep(0);
  wait_async0();
  __syncthreads();
  computeStep(1);

  int half = l >> 4, lc = l & 15;
#pragma unroll
  for (int nt = 0; nt < 4; ++nt) {
    int n = n0 + nt * 16 + lc;
    float bv = bias ? bias[n] : 0.f;
#pragma unroll
    for (int j = 0; j < 8; ++j) {
      int m = m0 + w * 16 + j + 8 * half;
      float vv = (acc[nt][j] + bv) * scale;
      long idx = cOff + (long)m * N + n;
      if (Cf) Cf[idx] = vv;
      if (Cb) Cb[idx] = (bf16)vv;
    }
  }
}

// ---------------------------------------------------------------------------
// Fused attention kernel. Grid: (ceil(N/16)=19, B). 128 threads = 4 waves.
// Per block: 16 n-rows, all 16 heads, K=64 keys.
//   phase 1: aff_w -> log  via WMMA over flattened (n,k) x p x h
//   phase 2: per-wave heads {w, w+4, w+8, w+12}:
//            aff = q.k^T (WMMA, B from pre-transposed kT, contiguous),
//            + logaff, softmax(k) in-wave,
//            out = attn @ kproj (WMMA, B = natural [k][g] layout),
//            epilogue v + relu(self + out + out_b)
// ---------------------------------------------------------------------------
#define ATTN_LD 72

__global__ void __launch_bounds__(128) fused_attn_kernel(
    const float* __restrict__ pos_emb,   // (B,N,64,64) f32
    const bf16*  __restrict__ pos_wt,    // (64,16)  [p][h]
    const float* __restrict__ pos_b,     // (16)
    const bf16*  __restrict__ qb,        // (B,N,D) bf16, pre-scaled by 1/8
    const bf16*  __restrict__ kT,        // (B,H,DG,KK) bf16  [d][kk]
    const bf16*  __restrict__ kproj,     // (B,H,KK,DG) bf16  [k][g]
    const float* __restrict__ self_f32,  // (B,N,D)
    const float* __restrict__ v,         // (B,N,D)
    const float* __restrict__ out_b,     // (D)
    float* __restrict__ out)             // (B,N,D)
{
  __shared__ __align__(16) bf16 s_posw[PED_ * H_];            // 2 KB
  __shared__ __align__(16) bf16 s_logaff[16 * KK_ * H_];      // 32 KB [n][k][h]
  __shared__ __align__(16) bf16 s_attn[4][16 * ATTN_LD];      // 9 KB  per-wave [m][k]

  int b  = blockIdx.y;
  int n0 = blockIdx.x * 16;
  int tid = threadIdx.x;
  int w = tid >> 5, l = tid & 31;
  int half = l >> 4, lc = l & 15;

  // stage pos_wt (64x16 bf16 = 2048B = 128 int4)
  ((int4*)s_posw)[tid] = ((const int4*)pos_wt)[tid];
  __syncthreads();

  // ---- phase 1: position affinity -> log, into LDS ----
  for (int t = w * 16; t < w * 16 + 16; ++t) {
    int flat0 = t * 16;
    floatx8 acc = floatx8{0.f};
#pragma unroll
    for (int c = 0; c < 2; ++c) {
      bf16x16 a;
      {
        int m = l & 15;
        int kh = (l >> 4) << 3;
        int flat = flat0 + m;
        int nl = flat >> 6;
        int kk = flat & 63;
        int row_n = n0 + nl; if (row_n > N_ - 1) row_n = N_ - 1;
        const float* src = pos_emb +
            (((long)(b * N_ + row_n) * KK_ + kk) * PED_) + c * 32 + kh;
#pragma unroll
        for (int i = 0; i < 16; i += 2) {
          int p = ((i >> 3) << 4) + (i & 7);
          a[i]     = (bf16)src[p];
          a[i + 1] = (bf16)src[p + 1];
        }
      }
      bf16x16 bfr = frag_b_rows(s_posw + (c * 32) * H_, H_);
      acc = __builtin_amdgcn_wmma_f32_16x16x32_bf16(
          false, a, false, bfr, (short)0, acc, false, false);
    }
    int h  = lc;
    int nl = flat0 >> 6;
    int k0 = (flat0 & 63) + 8 * half;
    float pb = pos_b[h];
#pragma unroll
    for (int j = 0; j < 8; ++j) {
      float aw = acc[j] + pb;
      aw = aw > 0.f ? aw : 0.f;
      float lg = __logf(fmaxf(aw, 1e-6f));
      s_logaff[((long)nl * KK_ + (k0 + j)) * H_ + h] = (bf16)lg;
    }
  }
  __syncthreads();

  // ---- phase 2: per-wave head loop ----
  bf16* myattn = s_attn[w];
#pragma unroll 1
  for (int hi = 0; hi < 4; ++hi) {
    int h = w + 4 * hi;
    const bf16* ktb = kT    + (long)(b * H_ + h) * DG_ * KK_;  // [d][kk]
    const bf16* kpb = kproj + (long)(b * H_ + h) * KK_ * DG_;  // [k][g]

    // aff = q . k^T   (q pre-scaled by 1/8)
    floatx8 acc[4] = {floatx8{0.f}, floatx8{0.f}, floatx8{0.f}, floatx8{0.f}};
#pragma unroll
    for (int c = 0; c < 2; ++c) {
      bf16x16 a;
      {
        int m = l & 15;
        int row = n0 + m; if (row > N_ - 1) row = N_ - 1;
        const bf16* src = qb + ((long)(b * N_ + row)) * D_ + h * DG_ + c * 32 + 8 * half;
#pragma unroll
        for (int i = 0; i < 16; i += 2) {
          int d = ((i >> 3) << 4) + (i & 7);
          a[i]     = src[d];
          a[i + 1] = src[d + 1];
        }
      }
#pragma unroll
      for (int kt = 0; kt < 4; ++kt) {
        bf16x16 bf_ = frag_b_rows(ktb + (long)(c * 32) * KK_ + kt * 16, KK_);
        acc[kt] = __builtin_amdgcn_wmma_f32_16x16x32_bf16(
            false, a, false, bf_, (short)0, acc[kt], false, false);
      }
    }

    // weighted = aff + logaff ; softmax over k (rows live in 16-lane halves)
    float wgt[4][8];
#pragma unroll
    for (int kt = 0; kt < 4; ++kt)
#pragma unroll
      for (int j = 0; j < 8; ++j) {
        int m = j + 8 * half;
        int k = kt * 16 + lc;
        wgt[kt][j] = acc[kt][j] + (float)s_logaff[((long)m * KK_ + k) * H_ + h];
      }
#pragma unroll
    for (int j = 0; j < 8; ++j) {
      float mx = fmaxf(fmaxf(wgt[0][j], wgt[1][j]), fmaxf(wgt[2][j], wgt[3][j]));
#pragma unroll
      for (int s = 1; s <= 8; s <<= 1) mx = fmaxf(mx, __shfl_xor(mx, s, 32));
      float sum = 0.f;
#pragma unroll
      for (int kt = 0; kt < 4; ++kt) {
        wgt[kt][j] = __expf(wgt[kt][j] - mx);
        sum += wgt[kt][j];
      }
#pragma unroll
      for (int s = 1; s <= 8; s <<= 1) sum += __shfl_xor(sum, s, 32);
      float inv = 1.f / sum;
#pragma unroll
      for (int kt = 0; kt < 4; ++kt) wgt[kt][j] *= inv;
    }

    // attn -> LDS (per-wave region), then A-fragments for out GEMM
#pragma unroll
    for (int kt = 0; kt < 4; ++kt)
#pragma unroll
      for (int j = 0; j < 8; ++j) {
        int m = j + 8 * half;
        int k = kt * 16 + lc;
        myattn[m * ATTN_LD + k] = (bf16)wgt[kt][j];
      }

    floatx8 oacc[4] = {floatx8{0.f}, floatx8{0.f}, floatx8{0.f}, floatx8{0.f}};
#pragma unroll
    for (int c = 0; c < 2; ++c) {
      bf16x16 a = frag_a_rowmajor(myattn + c * 32, ATTN_LD);
#pragma unroll
      for (int gt = 0; gt < 4; ++gt) {
        bf16x16 bf_ = frag_b_rows(kpb + (long)(c * 32) * DG_ + gt * 16, DG_);
        oacc[gt] = __builtin_amdgcn_wmma_f32_16x16x32_bf16(
            false, a, false, bf_, (short)0, oacc[gt], false, false);
      }
    }

    // epilogue: out = v + relu(self_feat + out + out_b)
#pragma unroll
    for (int gt = 0; gt < 4; ++gt) {
#pragma unroll
      for (int j = 0; j < 8; ++j) {
        int m = j + 8 * half;
        int row = n0 + m;
        if (row < N_) {
          int d = h * DG_ + gt * 16 + lc;
          long idx = ((long)(b * N_ + row)) * D_ + d;
          float o = oacc[gt][j] + out_b[d] + self_f32[idx];
          o = o > 0.f ? o : 0.f;
          out[idx] = v[idx] + o;
        }
      }
    }
  }
}

// ---------------------------------------------------------------------------
// host-side launch
// ---------------------------------------------------------------------------
extern "C" void kernel_launch(void* const* d_in, const int* in_sizes, int n_in,
                              void* d_out, int out_size, void* d_ws, size_t ws_size,
                              hipStream_t stream) {
  (void)in_sizes; (void)n_in; (void)out_size; (void)ws_size;
  const float* v_in    = (const float*)d_in[0];
  const float* pos_emb = (const float*)d_in[1];
  const float* self_w  = (const float*)d_in[2];
  const float* self_b  = (const float*)d_in[3];
  const float* q_w     = (const float*)d_in[4];
  const float* q_b     = (const float*)d_in[5];
  const float* k_w     = (const float*)d_in[6];
  const float* k_b     = (const float*)d_in[7];
  const float* pos_w   = (const float*)d_in[8];
  const float* pos_b   = (const float*)d_in[9];
  const float* out_w   = (const float*)d_in[10];
  const float* out_b   = (const float*)d_in[11];
  float* out = (float*)d_out;

  const long BN  = (long)B_ * N_;      // 9600
  const long BND = BN * D_;            // 9,830,400

  // workspace carve-up (256B aligned)
  char* ws = (char*)d_ws;
  size_t off = 0;
  auto carve = [&](size_t bytes) -> char* {
    char* p = ws + off;
    off = (off + bytes + 255) & ~(size_t)255;
    return p;
  };
  bf16*  v_bf     = (bf16*)carve(BND * 2);
  bf16*  self_wt  = (bf16*)carve((size_t)D_ * D_ * 2);
  bf16*  q_wt     = (bf16*)carve((size_t)D_ * D_ * 2);
  bf16*  k_wt     = (bf16*)carve((size_t)D_ * D_ * 2);
  bf16*  pos_wt   = (bf16*)carve((size_t)PED_ * H_ * 2);
  bf16*  out_wt   = (bf16*)carve((size_t)H_ * D_ * DG_ * 2);
  float* self_f32 = (float*)carve(BND * 4);
  bf16*  self_bf  = (bf16*)carve(BND * 2);
  bf16*  q_bf     = (bf16*)carve(BND * 2);
  float* k_f32    = (float*)carve((size_t)B_ * KK_ * D_ * 4);
  bf16*  kT       = (bf16*)carve((size_t)B_ * H_ * DG_ * KK_ * 2);
  bf16*  kproj_bf = (bf16*)carve((size_t)B_ * H_ * KK_ * DG_ * 2);

  // 1) v -> bf16
  cvt_bf16_kernel<<<(BND + 255) / 256, 256, 0, stream>>>(v_in, v_bf, BND);

  // 2) weight transposes (f32 -> bf16, (R,C)->(C,R))
  transpose_cvt_kernel<<<dim3(32, 32, 1), 256, 0, stream>>>(
      self_w, self_wt, D_, D_, D_, D_, 0, 0, 0, 0, 1);
  transpose_cvt_kernel<<<dim3(32, 32, 1), 256, 0, stream>>>(
      q_w, q_wt, D_, D_, D_, D_, 0, 0, 0, 0, 1);
  transpose_cvt_kernel<<<dim3(32, 32, 1), 256, 0, stream>>>(
      k_w, k_wt, D_, D_, D_, D_, 0, 0, 0, 0, 1);
  transpose_cvt_kernel<<<dim3(2, 1, 1), 256, 0, stream>>>(
      pos_w, pos_wt, H_, PED_, PED_, H_, 0, 0, 0, 0, 1);
  transpose_cvt_kernel<<<dim3(32, 2, H_), 256, 0, stream>>>(
      out_w, out_wt, DG_, D_, D_, DG_,
      0, (long)DG_ * D_, 0, (long)D_ * DG_, 1);

  // 3) self_feat = v @ self_w^T + self_b     (M=9600,N=1024,K=1024)
  gemm_bf16_big_kernel<<<dim3(D_ / 128, BN / 128, 1), 256, 0, stream>>>(
      v_bf, self_wt, self_b, self_f32, self_bf, (int)BN, D_, D_, 1.0f);

  // 4) q = (self_feat @ q_w^T + q_b) * (1/8)  -> bf16 only
  gemm_bf16_big_kernel<<<dim3(D_ / 128, BN / 128, 1), 256, 0, stream>>>(
      self_bf, q_wt, q_b, nullptr, q_bf, (int)BN, D_, D_, 0.125f);

  // 5) k = nongt_feat @ k_w^T + k_b   (batched over b: M=64) -> f32
  gemm_bf16_kernel<<<dim3(D_ / 64, 1, B_), 128, 0, stream>>>(
      self_bf, (long)N_ * D_, 0, k_wt, 0, 0, k_b,
      k_f32, nullptr, (long)KK_ * D_, 0, KK_, D_, D_, 1, 1.0f);

  // 6) kT[b,h] = transpose of k[b,:,h*64:(h+1)*64]  -> (DG,KK) bf16 [d][kk]
  transpose_cvt_kernel<<<dim3(2, 2, B_ * H_), 256, 0, stream>>>(
      k_f32, kT, KK_, DG_, D_, KK_,
      (long)KK_ * D_, (long)DG_, (long)H_ * DG_ * KK_, (long)DG_ * KK_, H_);

  // 7) kproj[b,h] = nongt_feat[b] @ out_w[h]^T  (64x64, K=1024) -> bf16 [k][g]
  gemm_bf16_kernel<<<dim3(1, 1, B_ * H_), 128, 0, stream>>>(
      self_bf, (long)N_ * D_, 0,
      out_wt, 0, (long)D_ * DG_, nullptr,
      nullptr, kproj_bf, (long)H_ * KK_ * DG_, (long)KK_ * DG_,
      KK_, DG_, D_, H_, 1.0f);

  // 8) fused position-affinity + attention + output projection + residual
  fused_attn_kernel<<<dim3((N_ + 15) / 16, B_, 1), 128, 0, stream>>>(
      pos_emb, pos_wt, pos_b, q_bf, kT, kproj_bf, self_f32, v_in, out_b, out);
}
